// MultiHeadAttention_34119220199496
// MI455X (gfx1250) — matrix operand order repaired and checked by reading, not compiled, not run on hardware
//
#include <hip/hip_runtime.h>

// ---------------------------------------------------------------------------
// Multi-head attention forward for MI455X (gfx1250, wave32, WMMA).
// B=2, S=2048, D=1024, H=16, hd=64.
// Phase 1: Q/K/V projections (f16 WMMA, f32 accum), outputs staged as f16.
// Phase 2: flash-attention per (b,h,q-block), online softmax, double-buffered
//          K/V staging via GLOBAL_LOAD_ASYNC_TO_LDS (ASYNCcnt) when available.
// ---------------------------------------------------------------------------

typedef __attribute__((ext_vector_type(16))) _Float16 v16h;
typedef __attribute__((ext_vector_type(8)))  _Float16 v8h;
typedef __attribute__((ext_vector_type(4)))  _Float16 v4h;
typedef __attribute__((ext_vector_type(8)))  float    v8f;

union V16U { v16h v; v8h h[2]; };

#define DEV __device__ __forceinline__
#define AS1 __attribute__((address_space(1)))
#define AS3 __attribute__((address_space(3)))

#if defined(__gfx1250__) && __has_builtin(__builtin_amdgcn_global_load_async_to_lds_b128)
#define HAVE_ASYNC_LDS 1
#else
#define HAVE_ASYNC_LDS 0
#endif

constexpr int B_ = 2;
constexpr int S_ = 2048;
constexpr int D_ = 1024;
constexpr int H_ = 16;
constexpr int HD = 64;
constexpr int M_ = B_ * S_;      // 4096 rows for the projections

// ---- address-space helpers for async-LDS builtins -------------------------
// The builtin expects (int4 AS1*, int4 AS3*, imm, imm). Generic LDS pointer:
// low 32 bits are the DS byte address (ISA 10.2 aperture mapping), so integer
// truncation yields a valid addrspace(3) pointer value; global pointers carry
// the same 64-bit value in AS1.
typedef int b128v __attribute__((vector_size(16)));

DEV AS1 b128v* glb_ptr(const void* p) { return (AS1 b128v*)(uintptr_t)p; }
DEV AS3 b128v* lds_ptr(void* p)       { return (AS3 b128v*)(unsigned)(uintptr_t)p; }

DEV void wait_async0() {
#if HAVE_ASYNC_LDS
#if __has_builtin(__builtin_amdgcn_s_wait_asynccnt)
  __builtin_amdgcn_s_wait_asynccnt(0);
#else
  asm volatile("s_wait_asynccnt 0" ::: "memory");
#endif
#endif
}

// ---- WMMA fragment loaders, CDNA5 wave32 layouts (ISA 7.12.2) -------------
// A (16x32, f16): lane L<16 holds row M=L, K = {0..7, 16..23}; L>=16: K = {8..15, 24..31}
DEV v16h load_a_frag(const _Float16* base, int ld, int lane) {
  const _Float16* r = base + (lane & 15) * ld + ((lane >> 4) << 3);
  V16U u;
  u.h[0] = *(const v8h*)(r);
  u.h[1] = *(const v8h*)(r + 16);
  return u.v;
}
// B (32x16, f16): lane L<16 holds col N=L, K = {0..7, 16..23}; L>=16: K = {8..15, 24..31}
// base addresses a tile stored as [n][k] (each n-row contiguous in k).
DEV v16h load_b_frag(const _Float16* base, int ld, int lane) {
  const _Float16* r = base + (lane & 15) * ld + ((lane >> 4) << 3);
  V16U u;
  u.h[0] = *(const v8h*)(r);
  u.h[1] = *(const v8h*)(r + 16);
  return u.v;
}
DEV v8f wmma16(v16h a, v16h b, v8f c) {
  return __builtin_amdgcn_wmma_f32_16x16x32_f16(false, a, false, b, (short)0, c,
                                                false, false);
}

// ---------------------------------------------------------------------------
// Projection GEMM: Y[m,e] = sum_d X[m,d] * W[e,d] + bias[e]   (X: M_ x D_)
// Stored as f16:  vtrans==0 -> out[b][h][s][dh]   (Q, K)
//                 vtrans==1 -> out[b][h][dh][s]   (V, transposed for PV GEMM)
// ---------------------------------------------------------------------------
constexpr int PT_M = 128, PT_N = 128, PT_K = 32;
constexpr int PLD  = 40;   // padded LDS row stride in halves (80B, 16B aligned)

__global__ __launch_bounds__(256, 2)
void mha_proj_kernel(const float* __restrict__ X, const float* __restrict__ W,
                     const float* __restrict__ bias, _Float16* __restrict__ out,
                     int vtrans, float scale)
{
  __shared__ _Float16 sA[PT_M * PLD];
  __shared__ _Float16 sB[PT_N * PLD];

  const int tid  = threadIdx.x;
  const int lane = tid & 31;
  const int wave = tid >> 5;          // 8 waves
  const int wm   = wave >> 1;         // 0..3 : 32-row group
  const int wn   = wave & 1;          // 0..1 : 64-col group
  const int bm   = blockIdx.x * PT_M;
  const int bn   = blockIdx.y * PT_N;

  const v8f zv = {};
  v8f acc[2][4];
#pragma unroll
  for (int mt = 0; mt < 2; ++mt)
#pragma unroll
    for (int nt = 0; nt < 4; ++nt) acc[mt][nt] = zv;

  for (int kk = 0; kk < D_; kk += PT_K) {
    // Stage 128x32 A and B tiles: fp32 global -> f16 LDS (coalesced float4)
#pragma unroll
    for (int p = 0; p < 4; ++p) {
      int idx = tid + p * 256;            // 1024 chunks of 4 floats
      int row = idx >> 3;                 // 8 chunks per 32-wide row
      int col = (idx & 7) << 2;
      float4 fa = *(const float4*)(X + (size_t)(bm + row) * D_ + kk + col);
      float4 fb = *(const float4*)(W + (size_t)(bn + row) * D_ + kk + col);
      v4h ha = { (_Float16)fa.x, (_Float16)fa.y, (_Float16)fa.z, (_Float16)fa.w };
      v4h hb = { (_Float16)fb.x, (_Float16)fb.y, (_Float16)fb.z, (_Float16)fb.w };
      *(v4h*)(sA + row * PLD + col) = ha;
      *(v4h*)(sB + row * PLD + col) = hb;
      if (kk + PT_K < D_) {               // prefetch next K tile (near scope)
        __builtin_prefetch(X + (size_t)(bm + row) * D_ + kk + PT_K + col, 0, 2);
        __builtin_prefetch(W + (size_t)(bn + row) * D_ + kk + PT_K + col, 0, 2);
      }
    }
    __syncthreads();

    v16h af[2], bf[4];
#pragma unroll
    for (int mt = 0; mt < 2; ++mt)
      af[mt] = load_a_frag(sA + (wm * 32 + mt * 16) * PLD, PLD, lane);
#pragma unroll
    for (int nt = 0; nt < 4; ++nt)
      bf[nt] = load_b_frag(sB + (wn * 64 + nt * 16) * PLD, PLD, lane);
#pragma unroll
    for (int mt = 0; mt < 2; ++mt)
#pragma unroll
      for (int nt = 0; nt < 4; ++nt)
        acc[mt][nt] = wmma16(af[mt], bf[nt], acc[mt][nt]);
    __syncthreads();
  }

  // Epilogue: bias, fold-in scale, scatter to head-major f16 layout.
  const int n    = lane & 15;
  const int moff = (lane >> 4) << 3;      // C layout: m = vgpr + 8*(lane>=16)
#pragma unroll
  for (int nt = 0; nt < 4; ++nt) {
    int e   = bn + wn * 64 + nt * 16 + n;
    float bb = bias[e];
    int hh  = e >> 6;                     // e / HD
    int dh  = e & (HD - 1);
#pragma unroll
    for (int mt = 0; mt < 2; ++mt) {
#pragma unroll
      for (int r = 0; r < 8; ++r) {
        int m  = bm + wm * 32 + mt * 16 + moff + r;
        int bi = m >> 11;                 // m / S_
        int s  = m & (S_ - 1);
        float val = (acc[mt][nt][r] + bb) * scale;
        size_t addr = vtrans
          ? ((((size_t)bi * H_ + hh) * HD + dh) * S_ + s)
          : ((((size_t)bi * H_ + hh) * S_ + s) * HD + dh);
        out[addr] = (_Float16)val;
      }
    }
  }
}

// ---------------------------------------------------------------------------
// Flash attention: one block per (b, h, 128 q rows); 4 waves x 32 q rows.
// qh: [B,H,S,hd] f16 (pre-scaled by 1/sqrt(hd)); kh: [B,H,S,hd]; vt: [B,H,hd,S]
// K/V blocks double-buffered; staged with async global->LDS DMA if available.
// ---------------------------------------------------------------------------
constexpr int QB  = 128;
constexpr int KBL = 32;
constexpr int LK  = 72;   // sK stride (halves)
constexpr int LV  = 40;   // sV stride
constexpr int LP  = 40;   // per-wave P-transpose scratch stride

DEV void stage_kv(_Float16* dK, _Float16* dV,
                  const _Float16* kbase, const _Float16* vbase,
                  int kb0, int tid)
{
#pragma unroll
  for (int p = 0; p < 2; ++p) {
    int c  = tid + p * 128;              // 256 chunks of 8 halves each
    int kr = c >> 3, kc = (c & 7) << 3;
    int vr = c >> 2, vc = (c & 3) << 3;
    const _Float16* gk = kbase + (size_t)(kb0 + kr) * HD + kc;
    const _Float16* gv = vbase + (size_t)vr * S_ + kb0 + vc;
    _Float16* lk = dK + kr * LK + kc;
    _Float16* lv = dV + vr * LV + vc;
#if HAVE_ASYNC_LDS
    __builtin_amdgcn_global_load_async_to_lds_b128(glb_ptr(gk), lds_ptr(lk), 0, 0);
    __builtin_amdgcn_global_load_async_to_lds_b128(glb_ptr(gv), lds_ptr(lv), 0, 0);
#else
    *(v8h*)lk = *(const v8h*)gk;
    *(v8h*)lv = *(const v8h*)gv;
#endif
  }
}

__global__ __launch_bounds__(128, 2)
void mha_flash_kernel(const _Float16* __restrict__ qh, const _Float16* __restrict__ kh,
                      const _Float16* __restrict__ vt, const unsigned char* __restrict__ mask,
                      float* __restrict__ out)
{
  __shared__ _Float16 sK[2][KBL * LK];     // [buf][kpos][d]
  __shared__ _Float16 sV[2][HD * LV];      // [buf][d][kpos]
  __shared__ _Float16 sP[4 * 32 * LP];     // per-wave C->A transpose scratch

  const int tid  = threadIdx.x;
  const int lane = tid & 31;
  const int wave = tid >> 5;               // 4 waves
  const int qb   = blockIdx.x * QB;
  const int h    = blockIdx.y;
  const int b    = blockIdx.z;

  const _Float16* qbase = qh + (((size_t)b * H_ + h) * S_ + qb + wave * 32) * HD;
  const _Float16* kbase = kh + ((size_t)b * H_ + h) * (size_t)S_ * HD;
  const _Float16* vbase = vt + ((size_t)b * H_ + h) * (size_t)HD * S_;
  const unsigned char* mbase = mask + (size_t)b * S_;

  const int n    = lane & 15;
  const int moff = (lane >> 4) << 3;

  // Q fragments resident for the whole pass: 2 m-tiles x 2 K-steps of hd=64
  v16h qf[2][2];
#pragma unroll
  for (int mt = 0; mt < 2; ++mt)
#pragma unroll
    for (int ks = 0; ks < 2; ++ks)
      qf[mt][ks] = load_a_frag(qbase + (size_t)mt * 16 * HD + ks * 32, HD, lane);

  const v8f zv = {};
  v8f o[2][4];
#pragma unroll
  for (int mt = 0; mt < 2; ++mt)
#pragma unroll
    for (int nt = 0; nt < 4; ++nt) o[mt][nt] = zv;

  float mrow[2][8], lrow[2][8];
#pragma unroll
  for (int mt = 0; mt < 2; ++mt)
#pragma unroll
    for (int r = 0; r < 8; ++r) { mrow[mt][r] = -1.0e30f; lrow[mt][r] = 0.0f; }

  // Prologue: stage block 0 into buffer 0.
  stage_kv(sK[0], sV[0], kbase, vbase, 0, tid);
  wait_async0();
  __syncthreads();

  int cur = 0;
  for (int kb0 = 0; kb0 < S_; kb0 += KBL) {
    // Kick off async staging of the NEXT block into the other buffer while
    // this block is consumed (ping-pong; previous barrier made nxt free).
    if (kb0 + KBL < S_)
      stage_kv(sK[cur ^ 1], sV[cur ^ 1], kbase, vbase, kb0 + KBL, tid);

    const _Float16* cK = sK[cur];
    const _Float16* cV = sV[cur];

    // S = Q * K^T  (Q pre-scaled by 1/8)
    v8f st[2][2];
    st[0][0] = zv; st[0][1] = zv; st[1][0] = zv; st[1][1] = zv;
#pragma unroll
    for (int ks = 0; ks < 2; ++ks) {
      v16h bk0 = load_b_frag(cK + 0  * LK + ks * 32, LK, lane);
      v16h bk1 = load_b_frag(cK + 16 * LK + ks * 32, LK, lane);
#pragma unroll
      for (int mt = 0; mt < 2; ++mt) {
        st[mt][0] = wmma16(qf[mt][ks], bk0, st[mt][0]);
        st[mt][1] = wmma16(qf[mt][ks], bk1, st[mt][1]);
      }
    }

    // Key-side mask (reference: scores = -inf where mask[b,k])
#pragma unroll
    for (int nt = 0; nt < 2; ++nt) {
      if (mbase[kb0 + nt * 16 + n]) {
#pragma unroll
        for (int mt = 0; mt < 2; ++mt)
#pragma unroll
          for (int r = 0; r < 8; ++r) st[mt][nt][r] = -3.0e38f;
      }
    }

    // Online softmax; row reductions across the 16 lanes of each lane-half.
#pragma unroll
    for (int mt = 0; mt < 2; ++mt) {
      float curm[8], al[8], rs[8];
#pragma unroll
      for (int r = 0; r < 8; ++r) curm[r] = fmaxf(st[mt][0][r], st[mt][1][r]);
#pragma unroll
      for (int off = 1; off < 16; off <<= 1)
#pragma unroll
        for (int r = 0; r < 8; ++r) curm[r] = fmaxf(curm[r], __shfl_xor(curm[r], off, 32));
#pragma unroll
      for (int r = 0; r < 8; ++r) {
        float nm = fmaxf(mrow[mt][r], curm[r]);
        al[r] = __expf(mrow[mt][r] - nm);
        mrow[mt][r] = nm;
      }
#pragma unroll
      for (int r = 0; r < 8; ++r) {
        float p0 = __expf(st[mt][0][r] - mrow[mt][r]);
        float p1 = __expf(st[mt][1][r] - mrow[mt][r]);
        st[mt][0][r] = p0; st[mt][1][r] = p1;
        rs[r] = p0 + p1;
      }
#pragma unroll
      for (int off = 1; off < 16; off <<= 1)
#pragma unroll
        for (int r = 0; r < 8; ++r) rs[r] += __shfl_xor(rs[r], off, 32);
#pragma unroll
      for (int r = 0; r < 8; ++r) lrow[mt][r] = lrow[mt][r] * al[r] + rs[r];
#pragma unroll
      for (int nt = 0; nt < 4; ++nt)
#pragma unroll
        for (int r = 0; r < 8; ++r) o[mt][nt][r] *= al[r];
    }

    // Transpose P from C layout to A layout via per-wave LDS scratch.
    _Float16* ps = sP + wave * 32 * LP;
#pragma unroll
    for (int mt = 0; mt < 2; ++mt)
#pragma unroll
      for (int nt = 0; nt < 2; ++nt)
#pragma unroll
        for (int r = 0; r < 8; ++r)
          ps[(mt * 16 + moff + r) * LP + nt * 16 + n] = (_Float16)st[mt][nt][r];
    asm volatile("s_wait_dscnt 0" ::: "memory");   // LDS write->read, same wave

    v16h pf0 = load_a_frag(ps,           LP, lane);
    v16h pf1 = load_a_frag(ps + 16 * LP, LP, lane);

    // O += P * V   (V^T staged: sV[d][kpos] -> B[k=kpos][n=d])
#pragma unroll
    for (int nt = 0; nt < 4; ++nt) {
      v16h bvf = load_b_frag(cV + nt * 16 * LV, LV, lane);
      o[0][nt] = wmma16(pf0, bvf, o[0][nt]);
      o[1][nt] = wmma16(pf1, bvf, o[1][nt]);
    }

    // Next block's DMA must be complete (and everyone done with cur) before
    // the buffers flip.
    wait_async0();
    __syncthreads();
    cur ^= 1;
  }

  // Normalize and store fp32 output [B, S, D].
#pragma unroll
  for (int mt = 0; mt < 2; ++mt) {
    float inv[8];
#pragma unroll
    for (int r = 0; r < 8; ++r) inv[r] = 1.0f / lrow[mt][r];
#pragma unroll
    for (int nt = 0; nt < 4; ++nt)
#pragma unroll
      for (int r = 0; r < 8; ++r) {
        int s = qb + wave * 32 + mt * 16 + moff + r;
        int e = h * HD + nt * 16 + n;
        out[((size_t)b * S_ + s) * D_ + e] = o[mt][nt][r] * inv[r];
      }
  }
}

// ---------------------------------------------------------------------------
extern "C" void kernel_launch(void* const* d_in, const int* in_sizes, int n_in,
                              void* d_out, int out_size, void* d_ws, size_t ws_size,
                              hipStream_t stream) {
  (void)in_sizes; (void)n_in; (void)out_size; (void)ws_size;
  const float* v  = (const float*)d_in[0];
  const float* k  = (const float*)d_in[1];
  const float* q  = (const float*)d_in[2];
  const unsigned char* mask = (const unsigned char*)d_in[3];
  const float* Wq = (const float*)d_in[4];
  const float* bq = (const float*)d_in[5];
  const float* Wk = (const float*)d_in[6];
  const float* bk = (const float*)d_in[7];
  const float* Wv = (const float*)d_in[8];
  const float* bv = (const float*)d_in[9];
  float* out = (float*)d_out;

  const size_t per = (size_t)B_ * H_ * S_ * HD;   // 4M halves = 8 MB each
  _Float16* qh = (_Float16*)d_ws;
  _Float16* kh = qh + per;
  _Float16* vt = kh + per;

  dim3 pgrid(M_ / PT_M, D_ / PT_N);               // (32, 8)
  const float inv_sqrt_hd = 0.125f;               // 1/sqrt(64), folded into Q
  mha_proj_kernel<<<pgrid, 256, 0, stream>>>(q, Wq, bq, qh, 0, inv_sqrt_hd);
  mha_proj_kernel<<<pgrid, 256, 0, stream>>>(k, Wk, bk, kh, 0, 1.0f);
  mha_proj_kernel<<<pgrid, 256, 0, stream>>>(v, Wv, bv, vt, 1, 1.0f);

  dim3 agrid(S_ / QB, H_, B_);                    // (16, 16, 2)
  mha_flash_kernel<<<agrid, 128, 0, stream>>>(qh, kh, vt, mask, out);
}
